// WindowAttention_2508260901419
// MI455X (gfx1250) — matrix operand order repaired and checked by reading, not compiled, not run on hardware
//
#include <hip/hip_runtime.h>
#include <hip/hip_bf16.h>
#include <math.h>

// ---------------------------------------------------------------------------
// Types for WMMA fragments (wave32, gfx1250)
// ---------------------------------------------------------------------------
typedef _Float16 h8  __attribute__((ext_vector_type(8)));
typedef _Float16 v16h __attribute__((ext_vector_type(16)));
typedef float    v8f  __attribute__((ext_vector_type(8)));

static __device__ __forceinline__ v16h cat16(h8 a, h8 b) {
    return __builtin_shufflevector(a, b, 0,1,2,3,4,5,6,7,8,9,10,11,12,13,14,15);
}

static __device__ __forceinline__ v8f wmma16(v16h a, v16h b, v8f c) {
    // D = A(16x32 f16) * B(32x16 f16) + C(16x16 f32)
    return __builtin_amdgcn_wmma_f32_16x16x32_f16(
        /*neg_a=*/false, a, /*neg_b=*/false, b,
        /*c_mod=*/(short)0, c, /*reuse_a=*/false, /*reuse_b=*/false);
}

// Fragment layouts (wave32):
//  A (16x32 f16): lane L -> row m = L&15, hi = L>>4.
//    elements 0..7  : K = kbase + 8*hi + e          (contiguous 8 halves)
//    elements 8..15 : K = kbase + 16 + 8*hi + (e-8) (contiguous 8 halves)
//  B (32x16 f16) loaded from B^T (row-major N x K): lane L -> col n = L&15.
//    elements 0..7 : K = kbase + 8*hi + e; 8..15 : K = kbase + 16 + 8*hi + (e-8) (same as A)
//  C/D (16x16 f32): VGPR r -> (M = r + 8*(L>>4), N = L&15)

// ---------------------------------------------------------------------------
// Problem constants
// ---------------------------------------------------------------------------
#define BATCH   64
#define NTOK    1024             // 32*32
#define DIM     256
#define INNER   512
#define HEADS   8
#define DHEAD   64
#define N3      1536             // 3*INNER
#define MROWS   (BATCH * NTOK)   // 65536
#define SCALE   0.125f           // 64^-0.5

// ---------------------------------------------------------------------------
// Elementwise conversion kernels
// ---------------------------------------------------------------------------
__global__ void cvt_f32_to_f16(const float* __restrict__ s, _Float16* __restrict__ d, int n) {
    int i = blockIdx.x * blockDim.x + threadIdx.x;
    if (i < n) d[i] = (_Float16)s[i];
}

// src: K x N (row-major f32)  ->  dst: N x K (row-major f16)  (i.e. transposed)
__global__ void transpose_to_f16(const float* __restrict__ s, _Float16* __restrict__ d, int K, int N) {
    int i = blockIdx.x * blockDim.x + threadIdx.x;
    if (i >= K * N) return;
    int n = i / K, k = i - n * K;
    d[(size_t)n * K + k] = (_Float16)s[(size_t)k * N + n];
}

// ---------------------------------------------------------------------------
// WMMA GEMM: one wave computes a 32x64 tile of C = A(MxK) * (B^T)^T.
// Two A row-groups share the four B fragments -> 8 WMMAs per 32-wide k-step,
// halving B traffic per FLOP versus a 16x64 tile.
// MODE 0: exact GELU -> fp16 C          (Ch, ldc = N)
// MODE 1: + bias, scatter to q/k [b,h,tok,64] and vT [b,h,64,1024] (fp16)
// MODE 2: + bias -> fp32 C              (Cf, ldc = N)
// ---------------------------------------------------------------------------
template <int MODE>
__global__ __launch_bounds__(256) void gemm_wmma(
    const _Float16* __restrict__ A, const _Float16* __restrict__ Bt,
    _Float16* __restrict__ Ch, const float* __restrict__ bias,
    _Float16* __restrict__ qb, _Float16* __restrict__ kb, _Float16* __restrict__ vb,
    float* __restrict__ Cf, int M, int N, int K)
{
    const int wid  = (blockIdx.x * blockDim.x + threadIdx.x) >> 5;
    const int lane = threadIdx.x & 31;
    const int ln = lane & 15, hi = lane >> 4;

    const int mtiles = M >> 5, ntiles = N >> 6;
    if (wid >= mtiles * ntiles) return;
    const int mt = wid / ntiles, nt = wid - mt * ntiles;
    const int mbase = mt << 5, nbase = nt << 6;

    v8f acc[2][4] = {};
    const _Float16* arow0 = A + (size_t)(mbase + ln) * K;
    const _Float16* arow1 = A + (size_t)(mbase + 16 + ln) * K;

    for (int k = 0; k < K; k += 32) {
        h8 a0lo = *(const h8*)(arow0 + k + 8 * hi);
        h8 a0hi = *(const h8*)(arow0 + k + 16 + 8 * hi);
        h8 a1lo = *(const h8*)(arow1 + k + 8 * hi);
        h8 a1hi = *(const h8*)(arow1 + k + 16 + 8 * hi);
        v16h af0 = cat16(a0lo, a0hi);
        v16h af1 = cat16(a1lo, a1hi);
#pragma unroll
        for (int t = 0; t < 4; ++t) {
            const _Float16* brow = Bt + (size_t)(nbase + t * 16 + ln) * K + k + 8 * hi;
            h8 blo = *(const h8*)(brow);
            h8 bhi = *(const h8*)(brow + 16);
            v16h bf = cat16(blo, bhi);
            acc[0][t] = wmma16(af0, bf, acc[0][t]);
            acc[1][t] = wmma16(af1, bf, acc[1][t]);
        }
    }

    // Epilogue: lane holds D(r, ln): rows mbase + 16g + r + 8*hi, cols nbase + 16t + ln
#pragma unroll
    for (int g = 0; g < 2; ++g) {
#pragma unroll
        for (int r = 0; r < 8; ++r) {
            const int mm = mbase + g * 16 + r + 8 * hi;
#pragma unroll
            for (int t = 0; t < 4; ++t) {
                const int nn = nbase + t * 16 + ln;
                float v = acc[g][t][r];
                if (MODE == 0) {
                    // exact GELU: 0.5*x*(1+erf(x/sqrt(2)))
                    float gl = 0.5f * v * (1.0f + erff(v * 0.70710678118654752f));
                    Ch[(size_t)mm * N + nn] = (_Float16)gl;
                } else if (MODE == 1) {
                    float val = v + bias[nn];
                    _Float16 hv = (_Float16)val;
                    const int b = mm >> 10, tok = mm & 1023;
                    int j = nn;
                    if (j < INNER) {                       // q: [b,h,tok,64]
                        int hh = j >> 6, dd = j & 63;
                        qb[(((size_t)(b * HEADS + hh)) * NTOK + tok) * DHEAD + dd] = hv;
                    } else if (j < 2 * INNER) {            // k: [b,h,tok,64]
                        j -= INNER;
                        int hh = j >> 6, dd = j & 63;
                        kb[(((size_t)(b * HEADS + hh)) * NTOK + tok) * DHEAD + dd] = hv;
                    } else {                               // v transposed: [b,h,64,1024]
                        j -= 2 * INNER;
                        int hh = j >> 6, dd = j & 63;
                        vb[(((size_t)(b * HEADS + hh)) * DHEAD + dd) * NTOK + tok] = hv;
                    }
                } else {
                    Cf[(size_t)mm * N + nn] = v + bias[nn];
                }
            }
        }
    }
}

// ---------------------------------------------------------------------------
// Windowed attention. One wave handles one (window, 16-row q-tile, b, h).
// Windows: stride_sel 0 -> 16 full 8x8 windows; stride_sel 1 -> 25 shifted
// windows with edge sizes 4/8 (zero-padded keys, softmax over padding).
// ---------------------------------------------------------------------------
static __device__ __forceinline__ int tok_of(int s, int r0, int c0, int wsh, int wmask) {
    return ((r0 + (s >> wsh)) << 5) + c0 + (s & wmask);
}

__global__ __launch_bounds__(128) void attn_window(
    const _Float16* __restrict__ qb, const _Float16* __restrict__ kb,
    const _Float16* __restrict__ vtb, float* __restrict__ out,
    int stride_sel, int accumulate)
{
    __shared__ __align__(16) _Float16 Plds[4 * 16 * 64];   // per-wave 16x64 P tile
    const int warp = threadIdx.x >> 5;
    const int lane = threadIdx.x & 31;
    const int ln = lane & 15, hi = lane >> 4;
    const int wid = blockIdx.x * 4 + warp;

    const int h   = wid & 7;
    const int b   = (wid >> 3) & 63;
    const int mt  = (wid >> 9) & 3;
    const int win = wid >> 11;
    const int nwin = stride_sel ? 25 : 16;
    const int nseg = stride_sel ? 5 : 4;
    if (win >= nwin) return;

    const int si = win / nseg, sj = win - si * nseg;
    int r0, hgt, c0, wc;
    if (stride_sel == 0) { r0 = si << 3; hgt = 8; c0 = sj << 3; wc = 8; }
    else {
        r0 = si ? (si * 8 - 4) : 0; hgt = (si == 0 || si == 4) ? 4 : 8;
        c0 = sj ? (sj * 8 - 4) : 0; wc  = (sj == 0 || sj == 4) ? 4 : 8;
    }
    const int num = hgt * wc;            // valid tokens in window (16/32/64)
    if (mt * 16 >= num) return;
    const int wsh = (wc == 4) ? 2 : 3, wmask = wc - 1;

    const size_t hb = ((size_t)(b * HEADS + h)) * NTOK * DHEAD;

    // ---- S = (Q K^T) for a 16x64 score tile, padded key columns -> 0 ------
    const int mrow = mt * 16 + ln;
    const int mq = mrow < num ? mrow : (num - 1);
    const _Float16* qrow = qb + hb + (size_t)tok_of(mq, r0, c0, wsh, wmask) * DHEAD;

    v8f s[4] = {};
#pragma unroll
    for (int k = 0; k < DHEAD; k += 32) {
        h8 alo = *(const h8*)(qrow + k + 8 * hi);
        h8 ahi = *(const h8*)(qrow + k + 16 + 8 * hi);
        v16h af = cat16(alo, ahi);
#pragma unroll
        for (int t = 0; t < 4; ++t) {
            const int n = t * 16 + ln;
            const bool valid = n < num;
            const int tn = tok_of(valid ? n : 0, r0, c0, wsh, wmask);
            const _Float16* krow = kb + hb + (size_t)tn * DHEAD + k + 8 * hi;
            h8 blo = *(const h8*)(krow);
            h8 bhi = *(const h8*)(krow + 16);
            if (!valid) { blo = (h8){}; bhi = (h8){}; }   // zero-padded keys
            s[t] = wmma16(af, cat16(blo, bhi), s[t]);
        }
    }

    // ---- softmax over 64 columns (row = r + 8*hi, spread over 16 lanes) ---
    _Float16* Pw = Plds + warp * (16 * 64);
#pragma unroll
    for (int r = 0; r < 8; ++r) {
        float mx = -3.4e38f;
#pragma unroll
        for (int t = 0; t < 4; ++t) mx = fmaxf(mx, s[t][r] * SCALE);
        for (int msk = 1; msk < 16; msk <<= 1) mx = fmaxf(mx, __shfl_xor(mx, msk, 32));
        float p[4], sum = 0.0f;
#pragma unroll
        for (int t = 0; t < 4; ++t) { p[t] = expf(s[t][r] * SCALE - mx); sum += p[t]; }
        for (int msk = 1; msk < 16; msk <<= 1) sum += __shfl_xor(sum, msk, 32);
        const float inv = 1.0f / sum;
#pragma unroll
        for (int t = 0; t < 4; ++t) {
            const int n = t * 16 + ln;
            const float pv = (n < num) ? p[t] * inv : 0.0f;   // zero padded V rows
            Pw[(r + 8 * hi) * 64 + n] = (_Float16)pv;
        }
    }
    asm volatile("s_wait_dscnt 0" ::: "memory");   // same-wave LDS RAW fence

    // ---- O = P @ V  (P from LDS, V^T rows contiguous in global) -----------
    v8f o[4] = {};
#pragma unroll
    for (int k = 0; k < 64; k += 32) {
        const _Float16* pr = Pw + ln * 64 + k;
        h8 alo = *(const h8*)(pr + 8 * hi);
        h8 ahi = *(const h8*)(pr + 16 + 8 * hi);
        v16h af = cat16(alo, ahi);
#pragma unroll
        for (int t = 0; t < 4; ++t) {
            const int dd = t * 16 + ln;
            const _Float16* vrow = vtb + hb + (size_t)dd * NTOK;
            v16h bf = {};
#pragma unroll
            for (int e = 0; e < 16; ++e) {
                int kk = k + 8 * hi + (e < 8 ? e : e + 8);
                int ts = kk < num ? kk : 0;                 // P col is 0 there anyway
                bf[e] = vrow[tok_of(ts, r0, c0, wsh, wmask)];
            }
            o[t] = wmma16(af, bf, o[t]);
        }
    }

    // ---- store 0.5 * O into [b, token, h*64+dd] (fp32 accumulator) --------
#pragma unroll
    for (int r = 0; r < 8; ++r) {
        const int m = mt * 16 + r + 8 * hi;
        if (m < num) {
            const int tokm = tok_of(m, r0, c0, wsh, wmask);
            float* orow = out + ((size_t)(b * NTOK + tokm)) * INNER + h * DHEAD;
#pragma unroll
            for (int t = 0; t < 4; ++t) {
                const int dd = t * 16 + ln;
                float val = 0.5f * o[t][r];
                if (accumulate) val += orow[dd];
                orow[dd] = val;
            }
        }
    }
}

// ---------------------------------------------------------------------------
// Host-side launch
// ---------------------------------------------------------------------------
extern "C" void kernel_launch(void* const* d_in, const int* in_sizes, int n_in,
                              void* d_out, int out_size, void* d_ws, size_t ws_size,
                              hipStream_t stream) {
    const float* x     = (const float*)d_in[0];
    const float* w_qkv = (const float*)d_in[1];
    const float* w_pro = (const float*)d_in[2];
    const float* b_pro = (const float*)d_in[3];
    const float* w_out = (const float*)d_in[4];
    const float* b_out = (const float*)d_in[5];
    float* outp = (float*)d_out;

    char* ws = (char*)d_ws;
    // Workspace layout (256B-aligned offsets; attention buffers alias dead G1)
    _Float16* XH    = (_Float16*)(ws + 0);                         //  32.0 MB
    _Float16* WQKVT = (_Float16*)(ws + 33554432ull);               //   0.75 MB
    _Float16* WPROT = (_Float16*)(ws + 34340864ull);               //   4.5 MB
    _Float16* WOUTT = (_Float16*)(ws + 39059456ull);               //   0.25 MB
    _Float16* G1    = (_Float16*)(ws + 39321600ull);               // 192.0 MB
    float*    AACC  = (float*)   (ws + 39321600ull);               // aliases G1 (128 MB)
    _Float16* AH    = (_Float16*)(ws + 173539328ull);              // aliases G1 tail (64 MB)
    _Float16* Q     = (_Float16*)(ws + 240648192ull);              //  64 MB
    _Float16* K     = (_Float16*)(ws + 307757056ull);              //  64 MB
    _Float16* VT    = (_Float16*)(ws + 374865920ull);              //  64 MB
    (void)in_sizes; (void)n_in; (void)out_size; (void)ws_size;

    // 1) fp16 conversions / weight transposes
    cvt_f32_to_f16<<<(MROWS * DIM + 255) / 256, 256, 0, stream>>>(x, XH, MROWS * DIM);
    transpose_to_f16<<<(DIM * N3 + 255) / 256, 256, 0, stream>>>(w_qkv, WQKVT, DIM, N3);
    transpose_to_f16<<<(N3 * N3 + 255) / 256, 256, 0, stream>>>(w_pro, WPROT, N3, N3);
    transpose_to_f16<<<(INNER * DIM + 255) / 256, 256, 0, stream>>>(w_out, WOUTT, INNER, DIM);

    // 2) G1 = gelu(x @ w_qkv)        [65536 x 1536] fp16
    {
        int waves = (MROWS / 32) * (N3 / 64);
        gemm_wmma<0><<<waves / 8, 256, 0, stream>>>(XH, WQKVT, G1, nullptr,
            nullptr, nullptr, nullptr, nullptr, MROWS, N3, DIM);
    }
    // 3) qkv = G1 @ w_pro + b_pro, scatter to Q/K/VT (fp16)
    {
        int waves = (MROWS / 32) * (N3 / 64);
        gemm_wmma<1><<<waves / 8, 256, 0, stream>>>(G1, WPROT, nullptr, b_pro,
            Q, K, VT, nullptr, MROWS, N3, N3);
    }
    // 4) windowed attention, mean over strides {0,4} -> AACC fp32 [65536 x 512]
    attn_window<<<(16 << 11) / 4, 128, 0, stream>>>(Q, K, VT, AACC, 0, 0);
    attn_window<<<(25 << 11) / 4, 128, 0, stream>>>(Q, K, VT, AACC, 1, 1);

    // 5) AACC -> fp16
    cvt_f32_to_f16<<<(MROWS * INNER + 255) / 256, 256, 0, stream>>>(AACC, AH, MROWS * INNER);

    // 6) out = AH @ w_out + b_out    [65536 x 256] fp32
    {
        int waves = (MROWS / 32) * (DIM / 64);
        gemm_wmma<2><<<waves / 8, 256, 0, stream>>>(AH, WOUTT, nullptr, b_out,
            nullptr, nullptr, nullptr, outp, MROWS, DIM, INNER);
    }
}
